// PCD_Corr_Align_45380624449692
// MI455X (gfx1250) — compile-verified, matrix operand order checked by reading, not run on hardware
//
#include <hip/hip_runtime.h>
#include <math.h>

typedef __attribute__((ext_vector_type(16))) __bf16 v16bf;
typedef __attribute__((ext_vector_type(8)))  float  v8f;

#define NFC 64
#define MAX_KPAD 160   // ceil((64+81)/32)*32

static __host__ __device__ inline int cdiv(int a, int b) { return (a + b - 1) / b; }
__device__ inline float lrelu_f(float v) { return v >= 0.f ? v : 0.1f * v; }

// ---------------------------------------------------------------------------
// zero the 4 absmean slots (graph-replay safe re-init)
__global__ void k_zero4(float* p) { if (threadIdx.x < 4) p[threadIdx.x] = 0.f; }

// ---------------------------------------------------------------------------
// bilinear resize, NCHW f32 -> NCHW f32 (matches reference half-pixel+clip math)
__global__ void k_resize(const float* __restrict__ in, float* __restrict__ out,
                         int B, int C, int Hi, int Wi, int Ho, int Wo, int total) {
  int idx = blockIdx.x * 256 + threadIdx.x;
  if (idx >= total) return;
  int ox = idx % Wo; int t = idx / Wo;
  int oy = t % Ho;   t /= Ho;
  int c  = t % C;    int b = t / C;
  float cy = ((float)oy + 0.5f) * ((float)Hi / (float)Ho) - 0.5f;
  cy = fminf(fmaxf(cy, 0.f), (float)(Hi - 1));
  int y0 = (int)floorf(cy); int y1 = min(y0 + 1, Hi - 1); float wy = cy - (float)y0;
  float cx = ((float)ox + 0.5f) * ((float)Wi / (float)Wo) - 0.5f;
  cx = fminf(fmaxf(cx, 0.f), (float)(Wi - 1));
  int x0 = (int)floorf(cx); int x1 = min(x0 + 1, Wi - 1); float wx = cx - (float)x0;
  const float* p = in + ((size_t)b * C + c) * Hi * Wi;
  float v = (p[(size_t)y0 * Wi + x0] * (1.f - wy) + p[(size_t)y1 * Wi + x0] * wy) * (1.f - wx)
          + (p[(size_t)y0 * Wi + x1] * (1.f - wy) + p[(size_t)y1 * Wi + x1] * wy) * wx;
  out[idx] = v;
}

// ---------------------------------------------------------------------------
// correlation (81 shifts, mean over 64 ch) + fused lrelu.  out: (B,81,H,W)
__global__ void k_corr(const float* __restrict__ a, const float* __restrict__ bb,
                       float* __restrict__ out, int B, int H, int W, int total) {
  int idx = blockIdx.x * 256 + threadIdx.x;
  if (idx >= total) return;
  int x = idx % W; int t = idx / W;
  int y = t % H;   t /= H;
  int d = t % 81;  int b = t / 81;
  int dy = d / 9 - 4, dx = d % 9 - 4;
  int yy = y + dy, xx = x + dx;
  float sum = 0.f;
  if (yy >= 0 && yy < H && xx >= 0 && xx < W) {
    size_t hw = (size_t)H * W;
    const float* pa = a  + (size_t)b * NFC * hw + (size_t)y * W + x;
    const float* pb = bb + (size_t)b * NFC * hw + (size_t)yy * W + xx;
    for (int c = 0; c < NFC; c++) sum += pa[c * hw] * pb[c * hw];
  }
  sum *= (1.f / (float)NFC);
  out[idx] = lrelu_f(sum);
}

// ---------------------------------------------------------------------------
// fused [nearest-up(cv,s) || ref] concat  ->  NHWC bf16, channels padded to Kpad
__global__ void k_cvt_corrref(const float* __restrict__ cv, const float* __restrict__ ref,
                              __bf16* __restrict__ out, int B, int H, int W, int s,
                              int Kpad, int total) {
  int idx = blockIdx.x * 256 + threadIdx.x;
  if (idx >= total) return;
  int ci = idx % Kpad; int t = idx / Kpad;
  int w  = t % W;      t /= W;
  int h  = t % H;      int b = t / H;
  float v = 0.f;
  if (ci < 81) {
    int Hs = H / s, Ws = W / s;
    v = cv[(((size_t)b * 81 + ci) * Hs + h / s) * Ws + w / s];
  } else if (ci < 81 + NFC) {
    v = ref[(((size_t)b * NFC + (ci - 81)) * H + h) * W + w];
  }
  out[idx] = (__bf16)v;
}

// ---------------------------------------------------------------------------
// [a || scaleB*b2] concat (Cb may be 0) NCHW f32 -> NHWC bf16, pad to Kpad
__global__ void k_cvt2(const float* __restrict__ a, int Ca,
                       const float* __restrict__ b2, int Cb, float scaleB,
                       __bf16* __restrict__ out, int B, int H, int W,
                       int Kpad, int total) {
  int idx = blockIdx.x * 256 + threadIdx.x;
  if (idx >= total) return;
  int ci = idx % Kpad; int t = idx / Kpad;
  int w  = t % W;      t /= W;
  int h  = t % H;      int b = t / H;
  float v = 0.f;
  if (ci < Ca)            v = a[(((size_t)b * Ca + ci) * H + h) * W + w];
  else if (ci < Ca + Cb)  v = scaleB * b2[(((size_t)b * Cb + (ci - Ca)) * H + h) * W + w];
  out[idx] = (__bf16)v;
}

// ---------------------------------------------------------------------------
// prepack OIHW f32 weights into per-lane WMMA B-fragments (bf16), zero-padded
// to Kpad x (NT*16).  layout: (((tap*KC + kc)*NT + nt)*32 + lane)*16 + j
__global__ void k_pack_w(const float* __restrict__ w, __bf16* __restrict__ out,
                         int Ci, int Co, int Kpad, int NT, int total) {
  int idx = blockIdx.x * 256 + threadIdx.x;
  if (idx >= total) return;
  int KC = Kpad >> 5;
  int j    = idx & 15;
  int lane = (idx >> 4) & 31;
  int r    = idx >> 9;
  int nt = r % NT;  r /= NT;
  int kc = r % KC;  int tap = r / KC;
  int n  = nt * 16 + (lane & 15);
  int hh = lane >> 4;
  int kk = (j < 8) ? (hh * 8 + j) : (16 + hh * 8 + (j - 8));
  int ci = kc * 32 + kk;
  int kh = tap / 3, kw = tap % 3;
  float v = 0.f;
  if (n < Co && ci < Ci) v = w[(((size_t)n * Ci + ci) * 3 + kh) * 3 + kw];
  out[idx] = (__bf16)v;
}

// ---------------------------------------------------------------------------
// implicit-GEMM 3x3 conv, pad=1, bf16 WMMA, f32 accumulate, fused bias(+lrelu).
// x: NHWC bf16 (B,H,W,Kpad);  out: NCHW f32 (B,Co,H,W).  W % 16 == 0.
// block = 128 threads (4 waves); M-tile = 16 pixels along W.
// UCNT = N-tiles per wave (compile-time, N padded to UCNT*4*16 channels), so
// every wave runs straight-line unconditional WMMAs (EXEC all-ones, acc in
// fixed VGPRs, no divergent guards).
// Input strip staged global->LDS via async-LDS loads (ASYNCcnt path).
template <int UCNT>
__global__ __launch_bounds__(128)
void k_conv3x3_wmma(const __bf16* __restrict__ xin, const __bf16* __restrict__ wpk,
                    const float* __restrict__ bias, float* __restrict__ out,
                    int B, int H, int W, int Kpad, int Co, int relu) {
  constexpr int NT = UCNT * 4;
  __shared__ __bf16 lds[3 * 18 * MAX_KPAD];
  const int ow0 = blockIdx.x * 16, oh = blockIdx.y, b = blockIdx.z;
  const int tid = threadIdx.x, lane = tid & 31, wave = tid >> 5;
  const int KC = Kpad >> 5;

  // stage 3x18xKpad input strip (zero-padded borders); in-bounds chunks use
  // global_load_async_to_lds_b128 (per-lane 16B, tracked with ASYNCcnt)
  const int vecs = Kpad >> 3;                       // 8 bf16 = 16B chunks / site
  for (int ch = tid; ch < 54 * vecs; ch += 128) {
    int site = ch / vecs, v = ch - site * vecs;
    int r = site / 18, c = site - r * 18;
    int ih = oh - 1 + r, iw = ow0 - 1 + c;
    __bf16* lp = &lds[(size_t)site * Kpad + v * 8];
    if (ih >= 0 && ih < H && iw >= 0 && iw < W) {
      unsigned lo = (unsigned)(size_t)lp;   // low 32 bits of shared addr = LDS offset
      unsigned long long ga =
          (unsigned long long)(xin + (((size_t)b * H + ih) * W + iw) * Kpad + v * 8);
      asm volatile("global_load_async_to_lds_b128 %0, %1, off"
                   :: "v"(lo), "v"(ga) : "memory");
    } else {
      *(uint4*)lp = make_uint4(0u, 0u, 0u, 0u);
    }
  }
  asm volatile("s_wait_asynccnt 0x0" ::: "memory");
  __syncthreads();

  v8f acc[UCNT];
#pragma unroll
  for (int u = 0; u < UCNT; u++)
#pragma unroll
    for (int r = 0; r < 8; r++) acc[u][r] = 0.f;

  union Frag { v16bf v; uint4 q[2]; };
  const int m = lane & 15, hh = lane >> 4;

  for (int tap = 0; tap < 9; ++tap) {
    const int kh = tap / 3, kw = tap % 3;
    for (int kc = 0; kc < KC; ++kc) {
      Frag a;                                       // A frag: 2x ds_load_b128
      int base = (kh * 18 + (m + kw)) * Kpad + kc * 32 + hh * 8;
      a.q[0] = *(const uint4*)(&lds[base]);
      a.q[1] = *(const uint4*)(&lds[base + 16]);
      const __bf16* wp = wpk + ((((size_t)tap * KC + kc) * NT) * 32 + lane) * 16
                       + (size_t)wave * 512;
#pragma unroll
      for (int u = 0; u < UCNT; ++u) {              // unconditional, acc in regs
        Frag wv;                                    // B frag: 2x global b128
        const __bf16* p = wp + (size_t)u * 4 * 512; // nt = wave + 4u
        wv.q[0] = *(const uint4*)(p);
        wv.q[1] = *(const uint4*)(p + 8);
        acc[u] = __builtin_amdgcn_wmma_f32_16x16x32_bf16(
            false, a.v, false, wv.v, (short)0, acc[u], false, false);
      }
    }
  }

  // store: lane -> column n, 8 contiguous pixels along W (mask pad channels)
#pragma unroll
  for (int u = 0; u < UCNT; ++u) {
    const int nt = wave + u * 4;
    int n = nt * 16 + m;
    if (n < Co) {
      float bv = bias[n];
      float* po = out + (((size_t)b * Co + n) * H + oh) * W + ow0 + hh * 8;
#pragma unroll
      for (int r = 0; r < 8; r++) {
        float v = acc[u][r] + bv;
        po[r] = relu ? lrelu_f(v) : v;
      }
    }
  }
}

// ---------------------------------------------------------------------------
// mean(|raw[:, 0:144]|) block-reduced into *slot
__global__ void k_absmean(const float* __restrict__ raw, float* __restrict__ slot,
                          int H, int W, int total, float inv) {
  __shared__ float red[256];
  int idx = blockIdx.x * 256 + threadIdx.x;
  float v = 0.f;
  if (idx < total) {
    size_t hw = (size_t)H * W;
    int p = idx % (int)hw;
    int c = (idx / (int)hw) % 144;
    int b = idx / ((int)hw * 144);
    v = fabsf(raw[((size_t)b * 216 + c) * hw + p]);
  }
  red[threadIdx.x] = v; __syncthreads();
  for (int s = 128; s > 0; s >>= 1) {
    if (threadIdx.x < s) red[threadIdx.x] += red[threadIdx.x + s];
    __syncthreads();
  }
  if (threadIdx.x == 0) atomicAdd(slot, red[0] * inv);
}

// ---------------------------------------------------------------------------
// deformable conv gather+accumulate. One block per pixel, thread = out channel,
// sampling channel == tid (== g*8+c). raw: (B,216,H,W) pre-activation offsets.
__global__ __launch_bounds__(64)
void k_dcn(const float* __restrict__ x, const float* __restrict__ raw,
           const float* __restrict__ w, const float* __restrict__ bias,
           float* __restrict__ out, int B, int H, int W, int relu) {
  __shared__ float s[64];
  const int px = blockIdx.x, py = blockIdx.y, b = blockIdx.z;
  const int t = threadIdx.x;
  const int g = t >> 3;
  const size_t hw = (size_t)H * W;
  const size_t pix = (size_t)py * W + px;
  const float* rawb = raw + (size_t)b * 216 * hw;
  const float* xc = x + ((size_t)b * NFC + t) * hw;
  float acc = bias[t];
  for (int k = 0; k < 9; k++) {
    int kh = k / 3, kw = k % 3;
    int i = g * 9 + k;
    float dy = rawb[(size_t)(2 * i) * hw + pix];
    float dx = rawb[(size_t)(2 * i + 1) * hw + pix];
    float mz = rawb[(size_t)(144 + i) * hw + pix];
    float mask = 1.f / (1.f + __expf(-mz));
    float pyf = (float)py + (float)(kh - 1) + dy;
    float pxf = (float)px + (float)(kw - 1) + dx;
    float y0f = floorf(pyf), x0f = floorf(pxf);
    float wy = pyf - y0f, wx = pxf - x0f;
    int y0 = (int)y0f, x0 = (int)x0f;
    float v00 = (y0     >= 0 && y0     < H && x0     >= 0 && x0     < W) ? xc[(size_t)y0 * W + x0]           : 0.f;
    float v01 = (y0     >= 0 && y0     < H && x0 + 1 >= 0 && x0 + 1 < W) ? xc[(size_t)y0 * W + x0 + 1]       : 0.f;
    float v10 = (y0 + 1 >= 0 && y0 + 1 < H && x0     >= 0 && x0     < W) ? xc[(size_t)(y0 + 1) * W + x0]     : 0.f;
    float v11 = (y0 + 1 >= 0 && y0 + 1 < H && x0 + 1 >= 0 && x0 + 1 < W) ? xc[(size_t)(y0 + 1) * W + x0 + 1] : 0.f;
    float v = v00 * (1.f - wy) * (1.f - wx) + v01 * (1.f - wy) * wx
            + v10 * wy * (1.f - wx) + v11 * wy * wx;
    s[t] = v * mask;
    __syncthreads();
    const float* wk = w + (size_t)t * NFC * 9 + kh * 3 + kw;
    float a2 = 0.f;
    for (int j = 0; j < NFC; j++) a2 += s[j] * wk[(size_t)j * 9];
    acc += a2;
    __syncthreads();
  }
  if (relu) acc = lrelu_f(acc);
  out[((size_t)b * NFC + t) * hw + pix] = acc;
}

// ===========================================================================
extern "C" void kernel_launch(void* const* d_in, const int* in_sizes, int n_in,
                              void* d_out, int out_size, void* d_ws, size_t ws_size,
                              hipStream_t stream) {
  (void)in_sizes; (void)n_in; (void)out_size; (void)ws_size;
  const int B = 2, H1 = 192, W1 = 192, H2 = 96, W2 = 96, H3 = 48, W3 = 48;

  // ---- inputs -------------------------------------------------------------
  const float* ref1 = (const float*)d_in[3];
  const float* ref2 = (const float*)d_in[4];
  const float* ref3 = (const float*)d_in[5];
  const float* last1 = (const float*)d_in[6];
  const float* last2 = (const float*)d_in[7];
  const float* last3 = (const float*)d_in[8];
  const float* nbr1 = (const float*)d_in[0];
  const float* nbr2 = (const float*)d_in[1];
  const float* nbr3 = (const float*)d_in[2];
  // params (dict insertion order, each *_w then *_b)
  const float *l3_oc1_w = (const float*)d_in[9],  *l3_oc1_b = (const float*)d_in[10];
  const float *l3_dcno_w = (const float*)d_in[11], *l3_dcno_b = (const float*)d_in[12];
  const float *l3_dcn_w = (const float*)d_in[13], *l3_dcn_b = (const float*)d_in[14];
  const float *l2_oc1_w = (const float*)d_in[15], *l2_oc1_b = (const float*)d_in[16];
  const float *l2_dcno_w = (const float*)d_in[17], *l2_dcno_b = (const float*)d_in[18];
  const float *l2_dcn_w = (const float*)d_in[19], *l2_dcn_b = (const float*)d_in[20];
  const float *l1_oc1_w = (const float*)d_in[21], *l1_oc1_b = (const float*)d_in[22];
  const float *l1_dcno_w = (const float*)d_in[23], *l1_dcno_b = (const float*)d_in[24];
  const float *l1_dcn_w = (const float*)d_in[25], *l1_dcn_b = (const float*)d_in[26];
  const float *cas_oc1_w = (const float*)d_in[27], *cas_oc1_b = (const float*)d_in[28];
  const float *cas_dcno_w = (const float*)d_in[29], *cas_dcno_b = (const float*)d_in[30];
  const float *cas_dcn_w = (const float*)d_in[31], *cas_dcn_b = (const float*)d_in[32];
  const float *l3_oc2_w = (const float*)d_in[33], *l3_oc2_b = (const float*)d_in[34];
  const float *l2_oc2_w = (const float*)d_in[35], *l2_oc2_b = (const float*)d_in[36];
  const float *l2_oc3_w = (const float*)d_in[37], *l2_oc3_b = (const float*)d_in[38];
  const float *l2_fea_w = (const float*)d_in[39], *l2_fea_b = (const float*)d_in[40];
  const float *l1_oc2_w = (const float*)d_in[41], *l1_oc2_b = (const float*)d_in[42];
  const float *l1_oc3_w = (const float*)d_in[43], *l1_oc3_b = (const float*)d_in[44];
  const float *l1_fea_w = (const float*)d_in[45], *l1_fea_b = (const float*)d_in[46];
  const float *cas_oc2_w = (const float*)d_in[47], *cas_oc2_b = (const float*)d_in[48];

  // ---- workspace carve ----------------------------------------------------
  char* wsp = (char*)d_ws;
  auto alloc = [&](size_t bytes) -> void* {
    void* p = (void*)wsp; wsp += (bytes + 255) & ~(size_t)255; return p;
  };
  const size_t E1 = (size_t)B * NFC * H1 * W1;            // 4,718,592
  __bf16* XBF  = (__bf16*)alloc((size_t)B * H1 * W1 * MAX_KPAD * 2);
  __bf16* WPK  = (__bf16*)alloc(262144 * 2);
  float* RAW   = (float*)alloc((size_t)B * 216 * H1 * W1 * 4);
  float* T0    = (float*)alloc(E1 * 4);
  float* T1    = (float*)alloc(E1 * 4);
  float* UP    = (float*)alloc(E1 * 4);
  float* L3OFF = (float*)alloc((size_t)B * NFC * H3 * W3 * 4);
  float* L3FEA = (float*)alloc((size_t)B * NFC * H3 * W3 * 4);
  float* L2OFF = (float*)alloc((size_t)B * NFC * H2 * W2 * 4);
  float* L2FEA = (float*)alloc((size_t)B * NFC * H2 * W2 * 4);
  float* L1FEA = (float*)alloc(E1 * 4);
  float* CV    = (float*)alloc((size_t)B * 81 * 12 * 12 * 4);
  float* R12A  = (float*)alloc((size_t)B * NFC * 12 * 12 * 4);
  float* R12B  = (float*)alloc((size_t)B * NFC * 12 * 12 * 4);

  float* OUTM  = (float*)d_out;
  float* SLOTS = OUTM + E1;

  // ---- helpers ------------------------------------------------------------
  auto resize = [&](const float* in, float* out, int C, int Hi, int Wi, int Ho, int Wo) {
    int total = B * C * Ho * Wo;
    k_resize<<<cdiv(total, 256), 256, 0, stream>>>(in, out, B, C, Hi, Wi, Ho, Wo, total);
  };
  auto corr = [&](const float* a, const float* bb) {
    int total = B * 81 * 12 * 12;
    k_corr<<<cdiv(total, 256), 256, 0, stream>>>(a, bb, CV, B, 12, 12, total);
  };
  auto cvt_corrref = [&](const float* ref, int H, int W, int s) {
    int total = B * H * W * MAX_KPAD;
    k_cvt_corrref<<<cdiv(total, 256), 256, 0, stream>>>(CV, ref, XBF, B, H, W, s, MAX_KPAD, total);
    return MAX_KPAD;
  };
  auto cvt2 = [&](const float* a, int Ca, const float* b2, int Cb, float sc, int H, int W) {
    int Kpad = (Ca + Cb + 31) & ~31;
    int total = B * H * W * Kpad;
    k_cvt2<<<cdiv(total, 256), 256, 0, stream>>>(a, Ca, b2, Cb, sc, XBF, B, H, W, Kpad, total);
    return Kpad;
  };
  auto conv = [&](int Kpad, int H, int W, const float* w, const float* bias,
                  int Ci, int Co, float* out, int relu) {
    int NTr = cdiv(Co, 64) * 4;                 // N-tiles rounded to multiple of 4
    int total = 9 * (Kpad >> 5) * NTr * 512;
    k_pack_w<<<cdiv(total, 256), 256, 0, stream>>>(w, WPK, Ci, Co, Kpad, NTr, total);
    dim3 g(W / 16, H, B);
    if (NTr == 4)
      k_conv3x3_wmma<1><<<g, 128, 0, stream>>>(XBF, WPK, bias, out, B, H, W, Kpad, Co, relu);
    else
      k_conv3x3_wmma<4><<<g, 128, 0, stream>>>(XBF, WPK, bias, out, B, H, W, Kpad, Co, relu);
  };
  auto absmean = [&](int H, int W, int slot) {
    int total = B * 144 * H * W;
    k_absmean<<<cdiv(total, 256), 256, 0, stream>>>(RAW, SLOTS + slot, H, W, total, 1.f / (float)total);
  };
  auto dcn = [&](const float* x, const float* w, const float* bias, float* out,
                 int H, int W, int relu) {
    k_dcn<<<dim3(W, H, B), 64, 0, stream>>>(x, RAW, w, bias, out, B, H, W, relu);
  };

  k_zero4<<<1, 4, 0, stream>>>(SLOTS);

  // ---- level 3 (48x48) ----------------------------------------------------
  resize(ref3, R12A, NFC, H3, W3, 12, 12);
  resize(nbr3, R12B, NFC, H3, W3, 12, 12);
  corr(R12A, R12B);
  cvt_corrref(ref3, H3, W3, 4);
  conv(MAX_KPAD, H3, W3, l3_oc1_w, l3_oc1_b, 145, NFC, T0, 1);
  { int k = cvt2(T0, NFC, nullptr, 0, 1.f, H3, W3);
    conv(k, H3, W3, l3_oc2_w, l3_oc2_b, NFC, NFC, L3OFF, 1); }
  { int k = cvt2(L3OFF, NFC, nullptr, 0, 1.f, H3, W3);
    conv(k, H3, W3, l3_dcno_w, l3_dcno_b, NFC, 216, RAW, 0); }
  absmean(H3, W3, 0);
  dcn(last3, l3_dcn_w, l3_dcn_b, L3FEA, H3, W3, 1);

  // ---- level 2 (96x96) ----------------------------------------------------
  resize(ref2, R12A, NFC, H2, W2, 12, 12);
  resize(nbr2, R12B, NFC, H2, W2, 12, 12);
  corr(R12A, R12B);
  cvt_corrref(ref2, H2, W2, 8);
  conv(MAX_KPAD, H2, W2, l2_oc1_w, l2_oc1_b, 145, NFC, T0, 1);
  resize(L3OFF, UP, NFC, H3, W3, H2, W2);
  { int k = cvt2(T0, NFC, UP, NFC, 2.f, H2, W2);
    conv(k, H2, W2, l2_oc2_w, l2_oc2_b, 2 * NFC, NFC, T1, 1); }
  { int k = cvt2(T1, NFC, nullptr, 0, 1.f, H2, W2);
    conv(k, H2, W2, l2_oc3_w, l2_oc3_b, NFC, NFC, L2OFF, 1); }
  { int k = cvt2(L2OFF, NFC, nullptr, 0, 1.f, H2, W2);
    conv(k, H2, W2, l2_dcno_w, l2_dcno_b, NFC, 216, RAW, 0); }
  absmean(H2, W2, 1);
  dcn(last2, l2_dcn_w, l2_dcn_b, T0, H2, W2, 0);
  resize(L3FEA, UP, NFC, H3, W3, H2, W2);
  { int k = cvt2(T0, NFC, UP, NFC, 1.f, H2, W2);
    conv(k, H2, W2, l2_fea_w, l2_fea_b, 2 * NFC, NFC, L2FEA, 1); }

  // ---- level 1 (192x192) --------------------------------------------------
  resize(ref1, R12A, NFC, H1, W1, 12, 12);
  resize(nbr1, R12B, NFC, H1, W1, 12, 12);
  corr(R12A, R12B);
  cvt_corrref(ref1, H1, W1, 16);
  conv(MAX_KPAD, H1, W1, l1_oc1_w, l1_oc1_b, 145, NFC, T0, 1);
  resize(L2OFF, UP, NFC, H2, W2, H1, W1);
  { int k = cvt2(T0, NFC, UP, NFC, 2.f, H1, W1);
    conv(k, H1, W1, l1_oc2_w, l1_oc2_b, 2 * NFC, NFC, T1, 1); }
  { int k = cvt2(T1, NFC, nullptr, 0, 1.f, H1, W1);
    conv(k, H1, W1, l1_oc3_w, l1_oc3_b, NFC, NFC, T0, 1); }     // L1_off
  { int k = cvt2(T0, NFC, nullptr, 0, 1.f, H1, W1);
    conv(k, H1, W1, l1_dcno_w, l1_dcno_b, NFC, 216, RAW, 0); }
  absmean(H1, W1, 2);
  dcn(last1, l1_dcn_w, l1_dcn_b, T1, H1, W1, 0);
  resize(L2FEA, UP, NFC, H2, W2, H1, W1);
  { int k = cvt2(T1, NFC, UP, NFC, 1.f, H1, W1);
    conv(k, H1, W1, l1_fea_w, l1_fea_b, 2 * NFC, NFC, L1FEA, 0); }

  // ---- cascade ------------------------------------------------------------
  resize(ref1, R12A, NFC, H1, W1, 12, 12);
  resize(L1FEA, R12B, NFC, H1, W1, 12, 12);
  corr(R12A, R12B);
  cvt_corrref(ref1, H1, W1, 16);
  conv(MAX_KPAD, H1, W1, cas_oc1_w, cas_oc1_b, 145, NFC, T0, 1);
  { int k = cvt2(T0, NFC, nullptr, 0, 1.f, H1, W1);
    conv(k, H1, W1, cas_oc2_w, cas_oc2_b, NFC, NFC, T1, 1); }
  { int k = cvt2(T1, NFC, nullptr, 0, 1.f, H1, W1);
    conv(k, H1, W1, cas_dcno_w, cas_dcno_b, NFC, 216, RAW, 0); }
  absmean(H1, W1, 3);
  dcn(L1FEA, cas_dcn_w, cas_dcn_b, OUTM, H1, W1, 1);   // final lrelu fused
}
